// GlobalGraphNet_30915174596975
// MI455X (gfx1250) — matrix operand order repaired and checked by reading, not compiled, not run on hardware
//
#include <hip/hip_runtime.h>
#include <hip/hip_bf16.h>

#define N_NODES 38332
#define POI_LEN 38333
#define N_EDGES (N_NODES * 32)
#define M_TILES ((N_NODES + 15) / 16)      // 2396
#define N16     (M_TILES * 16)             // 38336
#define KPAD_IN 416                        // 403 padded to 13*32
#define NEG_SLOPE 0.01f

typedef __bf16 bf16_t;
typedef __attribute__((ext_vector_type(16))) __bf16 v16bf;
typedef __attribute__((ext_vector_type(8)))  __bf16 v8bf;
typedef __attribute__((ext_vector_type(8)))  float  v8f;

// ---------------- utility ----------------
__global__ void k_zero_i32(int* p, int n) {
  int i = blockIdx.x * blockDim.x + threadIdx.x;
  if (i < n) p[i] = 0;
}
__global__ void k_zero_f32(float* p, int n) {
  int i = blockIdx.x * blockDim.x + threadIdx.x;
  if (i < n) p[i] = 0.f;
}

// ---------------- CSR construction ----------------
__global__ void k_count(const int* __restrict__ dst, int* __restrict__ counts) {
  int e = blockIdx.x * blockDim.x + threadIdx.x;
  if (e < N_EDGES) atomicAdd(&counts[dst[e]], 1);
}

__global__ void k_dinv(const int* __restrict__ counts, float* __restrict__ dinv) {
  int n = blockIdx.x * blockDim.x + threadIdx.x;
  if (n < N_NODES) dinv[n] = rsqrtf((float)(counts[n] + 1));  // deg includes self loop
}

__global__ __launch_bounds__(1024)
void k_scan(const int* __restrict__ counts, int* __restrict__ rowptr,
            int* __restrict__ cursor, int n) {
  __shared__ int sm[1024];
  int tid = threadIdx.x;
  int per = (n + 1023) / 1024;
  int base = tid * per;
  int sum = 0;
  for (int i = 0; i < per; ++i) { int idx = base + i; if (idx < n) sum += counts[idx]; }
  sm[tid] = sum; __syncthreads();
  for (int off = 1; off < 1024; off <<= 1) {
    int v = (tid >= off) ? sm[tid - off] : 0;
    __syncthreads();
    sm[tid] += v;
    __syncthreads();
  }
  int run = (tid == 0) ? 0 : sm[tid - 1];
  for (int i = 0; i < per; ++i) {
    int idx = base + i;
    if (idx < n) { rowptr[idx] = run; cursor[idx] = run; run += counts[idx]; }
  }
  if (tid == 1023) rowptr[n] = sm[1023];
}

__global__ void k_scatter(const int* __restrict__ src, const int* __restrict__ dst,
                          int* __restrict__ cursor, int* __restrict__ col) {
  int e = blockIdx.x * blockDim.x + threadIdx.x;
  if (e < N_EDGES) {
    int d = dst[e];
    int pos = atomicAdd(&cursor[d], 1);
    col[pos] = src[e];
  }
}

// ---------------- weight prep (transpose + f32->bf16) ----------------
__global__ void k_prep_w(const float* __restrict__ W_in, const float* __restrict__ gcn_Ws,
                         bf16_t* __restrict__ Wt_in, bf16_t* __restrict__ Wt_l) {
  int idx = blockIdx.x * blockDim.x + threadIdx.x;
  if (idx < 64 * KPAD_IN) {
    int n = idx / KPAD_IN, k = idx % KPAD_IN;
    float v = (k < 403) ? W_in[k * 64 + n] : 0.f;
    Wt_in[idx] = (bf16_t)v;
  } else {
    int r = idx - 64 * KPAD_IN;
    if (r < 5 * 4096) {
      int l = r / 4096, rr = r % 4096, n = rr / 64, k = rr % 64;
      Wt_l[r] = (bf16_t)gcn_Ws[l * 4096 + k * 64 + n];
    }
  }
}

// ---------------- input feature gather: embed lookup -> bf16 (N16 x 416) ----------------
__global__ __launch_bounds__(128)
void k_gather_in(const float* __restrict__ x5, const float* __restrict__ poi,
                 const float* __restrict__ cat, bf16_t* __restrict__ fin) {
  const int row = blockIdx.x;
  bf16_t* dstr = fin + (size_t)row * KPAD_IN;
  if (row >= N_NODES) {
    for (int k = threadIdx.x; k < KPAD_IN; k += 128) dstr[k] = (bf16_t)0.f;
    return;
  }
  const int p = (int)x5[row * 5 + 0];
  const int c = (int)x5[row * 5 + 1];
  for (int k = threadIdx.x; k < KPAD_IN; k += 128) {
    float v;
    if      (k < 300) v = poi[p * 300 + k];
    else if (k < 400) v = cat[c * 100 + (k - 300)];
    else if (k < 403) v = x5[row * 5 + (k - 398)];
    else              v = 0.f;
    dstr[k] = (bf16_t)v;
  }
}

// ---------------- WMMA GEMM helpers ----------------
__device__ __forceinline__ v16bf cat16(v8bf lo, v8bf hi) {
  return __builtin_shufflevector(lo, hi, 0, 1, 2, 3, 4, 5, 6, 7,
                                 8, 9, 10, 11, 12, 13, 14, 15);
}

// ---------------- input GEMM: (Fin N16x416 bf16) @ (W_in^T 64x416 bf16) ----------------
// grid.x = M_TILES, block = 128 (4 waves; wave w -> 16-col N tile w)
__global__ __launch_bounds__(128)
void k_gemm_in(const bf16_t* __restrict__ Fin, const bf16_t* __restrict__ Wt,
               float* __restrict__ h) {
  const int wave = threadIdx.x >> 5;
  const int lane = threadIdx.x & 31;
  const int m_l  = lane & 15;
  const int half = lane >> 4;
  const int row  = blockIdx.x * 16 + m_l;
  const int ncol = wave * 16 + m_l;
  const v8bf* Arow = (const v8bf*)(Fin + (size_t)row * KPAD_IN);   // 52 chunks of 8
  const v8bf* Brow = (const v8bf*)(Wt + (size_t)ncol * KPAD_IN);
  v8f c = {};
  for (int ks = 0; ks < KPAD_IN / 32; ++ks) {
    const int q = ks * 4;   // chunk index base (32 bf16 = 4 chunks)
    // A layout (16-bit 16x32): VGPR0-3 -> K = half*8+0..7 ; VGPR4-7 -> K = 16+half*8+0..7
    v16bf a = cat16(Arow[q + half], Arow[q + 2 + half]);
    // B layout (32x16, col = lane&15): same K order as A (half*8+0..7, 16+half*8+0..7)
    v16bf b = cat16(Brow[q + half], Brow[q + 2 + half]);
    c = __builtin_amdgcn_wmma_f32_16x16x32_bf16(false, a, false, b, (short)0, c, false, false);
  }
#pragma unroll
  for (int j = 0; j < 8; ++j) {          // C: VGPR j -> M = j + 8*half, N = lane&15
    int grow = blockIdx.x * 16 + j + half * 8;
    h[grow * 64 + ncol] = c[j];
  }
}

// ---------------- hidden-layer GEMM: (feat N16x64 bf16) @ (W^T 64x64 bf16) ----------------
__global__ __launch_bounds__(128)
void k_gemm64(const bf16_t* __restrict__ Abf, const bf16_t* __restrict__ Wt,
              float* __restrict__ h) {
  const int wave = threadIdx.x >> 5;
  const int lane = threadIdx.x & 31;
  const int m_l  = lane & 15;
  const int half = lane >> 4;
  const int row  = blockIdx.x * 16 + m_l;
  const int ncol = wave * 16 + m_l;
  const v8bf* Arow = (const v8bf*)(Abf + (size_t)row * 64);   // 8 chunks of 8
  const v8bf* Brow = (const v8bf*)(Wt + (size_t)ncol * 64);
  v8f c = {};
#pragma unroll
  for (int ks = 0; ks < 2; ++ks) {
    const int q = ks * 4;
    v16bf a = cat16(Arow[q + half], Arow[q + 2 + half]);
    v16bf b = cat16(Brow[q + half], Brow[q + 2 + half]);
    c = __builtin_amdgcn_wmma_f32_16x16x32_bf16(false, a, false, b, (short)0, c, false, false);
  }
#pragma unroll
  for (int j = 0; j < 8; ++j) {
    int grow = blockIdx.x * 16 + j + half * 8;
    h[grow * 64 + ncol] = c[j];
  }
}

// ---------------- CSR aggregation over 64 channels + bias + activation ----------------
// grid.x = N16, block = 64 (thread = channel). mode 0: leaky(t); mode 1: leaky(t)+t
// Writes f32 (for final dot) and bf16 (feeds next WMMA GEMM).
__global__ __launch_bounds__(64)
void k_agg64(const float* __restrict__ h, const int* __restrict__ rowptr,
             const int* __restrict__ col, const float* __restrict__ dinv,
             const float* __restrict__ bias, float* __restrict__ out_f,
             bf16_t* __restrict__ out_b, int mode) {
  const int n = blockIdx.x;
  const int cch = threadIdx.x;
  if (n >= N_NODES) {                   // zero the padded tail rows deterministically
    out_f[n * 64 + cch] = 0.f;
    out_b[n * 64 + cch] = (bf16_t)0.f;
    return;
  }
  const int start = rowptr[n], end = rowptr[n + 1];
  float acc = 0.f;
  for (int e = start; e < end; ++e) {
    int s = col[e];
    if (e + 1 < end) __builtin_prefetch(&h[col[e + 1] * 64], 0, 1);  // global_prefetch_b8
    acc += h[s * 64 + cch] * dinv[s];
  }
  float di = dinv[n];
  float t = acc * di + h[n * 64 + cch] * di * di + bias[cch];
  float l = (t >= 0.f) ? t : NEG_SLOPE * t;
  float r = mode ? (l + t) : l;
  out_f[n * 64 + cch] = r;
  out_b[n * 64 + cch] = (bf16_t)r;
}

// ---------------- feat @ W_out (64 -> 1) ----------------
__global__ __launch_bounds__(64)
void k_dot_out(const float* __restrict__ feat, const float* __restrict__ Wo,
               float* __restrict__ h1) {
  __shared__ float red[64];
  const int n = blockIdx.x, t = threadIdx.x;
  red[t] = feat[n * 64 + t] * Wo[t];
  __syncthreads();
  for (int s = 32; s > 0; s >>= 1) {
    if (t < s) red[t] += red[t + s];
    __syncthreads();
  }
  if (t == 0) h1[n] = red[0];
}

// ---------------- scalar-channel aggregation + leaky ----------------
__global__ void k_agg1(const float* __restrict__ h1, const int* __restrict__ rowptr,
                       const int* __restrict__ col, const float* __restrict__ dinv,
                       const float* __restrict__ b_out, float* __restrict__ s) {
  int n = blockIdx.x * blockDim.x + threadIdx.x;
  if (n >= N_NODES) return;
  float acc = 0.f;
  int start = rowptr[n], end = rowptr[n + 1];
  for (int e = start; e < end; ++e) { int sidx = col[e]; acc += h1[sidx] * dinv[sidx]; }
  float di = dinv[n];
  float t = acc * di + h1[n] * di * di + b_out[0];
  s[n] = (t >= 0.f) ? t : NEG_SLOPE * t;
}

// ---------------- fc1: s(38332) @ W(38332x128), blocked partial + atomics ----------------
__global__ __launch_bounds__(128)
void k_fc1(const float* __restrict__ s, const float* __restrict__ W,
           float* __restrict__ hpre) {
  __shared__ float sv[512];
  const int i0 = blockIdx.x * 512;
  for (int i = threadIdx.x; i < 512; i += 128) {
    int gi = i0 + i;
    sv[i] = (gi < N_NODES) ? s[gi] : 0.f;
  }
  __syncthreads();
  const int j = threadIdx.x;
  float acc = 0.f;
  for (int ii = 0; ii < 512; ++ii) {
    int gi = i0 + ii;
    if (gi < N_NODES) acc += sv[ii] * W[gi * 128 + j];
  }
  atomicAdd(&hpre[j], acc);
}

__global__ __launch_bounds__(128)
void k_fc1_act(const float* __restrict__ hpre, const float* __restrict__ b1,
               float* __restrict__ hvec) {
  int t = threadIdx.x;
  float v = hpre[t] + b1[t];
  hvec[t] = v > 0.f ? v : 0.f;
}

// ---------------- fc2: h(128) @ W2(128x38333) + b -> relu -> out ----------------
__global__ __launch_bounds__(256)
void k_fc2(const float* __restrict__ hvec, const float* __restrict__ W2,
           const float* __restrict__ b2, float* __restrict__ out) {
  __shared__ float hs[128];
  int t = threadIdx.x;
  if (t < 128) hs[t] = hvec[t];
  __syncthreads();
  int j = blockIdx.x * 256 + t;
  if (j >= POI_LEN) return;
  float acc = b2[j];
  for (int k = 0; k < 128; ++k) acc += hs[k] * W2[k * POI_LEN + j];
  out[j] = acc > 0.f ? acc : 0.f;
}

// ---------------- launch ----------------
extern "C" void kernel_launch(void* const* d_in, const int* in_sizes, int n_in,
                              void* d_out, int out_size, void* d_ws, size_t ws_size,
                              hipStream_t stream) {
  const float* x5      = (const float*)d_in[0];
  const int*   edges   = (const int*)  d_in[1];
  const float* poi_emb = (const float*)d_in[2];
  const float* cat_emb = (const float*)d_in[3];
  const float* W_in    = (const float*)d_in[4];
  const float* b_in    = (const float*)d_in[5];
  const float* gcn_Ws  = (const float*)d_in[6];
  const float* gcn_bs  = (const float*)d_in[7];
  const float* W_out   = (const float*)d_in[8];
  const float* b_out   = (const float*)d_in[9];
  const float* fc1_W   = (const float*)d_in[10];
  const float* fc1_b   = (const float*)d_in[11];
  const float* fc2_W   = (const float*)d_in[12];
  const float* fc2_b   = (const float*)d_in[13];
  float* out = (float*)d_out;

  const int* src = edges;             // edge_index[0]
  const int* dst = edges + N_EDGES;   // edge_index[1]

  char* w = (char*)d_ws;
  auto alloc = [&](size_t bytes) -> void* {
    void* p = (void*)w;
    w += (bytes + 255) & ~(size_t)255;
    return p;
  };
  float*  h_a    = (float*) alloc((size_t)N16 * 64 * 4);   // GEMM output
  float*  feat_f = (float*) alloc((size_t)N16 * 64 * 4);   // aggregated features (f32)
  bf16_t* feat_b = (bf16_t*)alloc((size_t)N16 * 64 * 2);   // aggregated features (bf16)
  bf16_t* fin    = (bf16_t*)alloc((size_t)N16 * KPAD_IN * 2); // gathered input feats
  int*    counts = (int*)   alloc((size_t)N_NODES * 4);
  int*    rowptr = (int*)   alloc((size_t)(N_NODES + 1) * 4);
  int*    cursor = (int*)   alloc((size_t)N_NODES * 4);
  int*    col    = (int*)   alloc((size_t)N_EDGES * 4);
  float*  dinv   = (float*) alloc((size_t)N_NODES * 4);
  float*  h1     = (float*) alloc((size_t)N_NODES * 4);
  float*  svec   = (float*) alloc((size_t)N_NODES * 4);
  float*  hpre   = (float*) alloc(128 * 4);
  float*  hvec   = (float*) alloc(128 * 4);
  bf16_t* Wt_in  = (bf16_t*)alloc((size_t)64 * KPAD_IN * 2);
  bf16_t* Wt_l   = (bf16_t*)alloc((size_t)5 * 4096 * 2);

  // 1. CSR + degrees
  k_zero_i32<<<(N_NODES + 255) / 256, 256, 0, stream>>>(counts, N_NODES);
  k_count<<<(N_EDGES + 255) / 256, 256, 0, stream>>>(dst, counts);
  k_dinv<<<(N_NODES + 255) / 256, 256, 0, stream>>>(counts, dinv);
  k_scan<<<1, 1024, 0, stream>>>(counts, rowptr, cursor, N_NODES);
  k_scatter<<<(N_EDGES + 255) / 256, 256, 0, stream>>>(src, dst, cursor, col);

  // 2. weight prep + input feature gather (f32 -> padded bf16)
  {
    int tot = 64 * KPAD_IN + 5 * 4096;
    k_prep_w<<<(tot + 255) / 256, 256, 0, stream>>>(W_in, gcn_Ws, Wt_in, Wt_l);
  }
  k_gather_in<<<N16, 128, 0, stream>>>(x5, poi_emb, cat_emb, fin);

  // 3. input layer: WMMA GEMM, then aggregate + leaky
  k_gemm_in<<<M_TILES, 128, 0, stream>>>(fin, Wt_in, h_a);
  k_agg64<<<N16, 64, 0, stream>>>(h_a, rowptr, col, dinv, b_in, feat_f, feat_b, 0);

  // 4. five hidden GCN layers: feat = leaky(t) + t
  for (int l = 0; l < 5; ++l) {
    k_gemm64<<<M_TILES, 128, 0, stream>>>(feat_b, Wt_l + (size_t)l * 4096, h_a);
    k_agg64<<<N16, 64, 0, stream>>>(h_a, rowptr, col, dinv, gcn_bs + l * 64,
                                    feat_f, feat_b, 1);
  }

  // 5. output conv (64 -> 1) + scalar aggregation + leaky
  k_dot_out<<<N_NODES, 64, 0, stream>>>(feat_f, W_out, h1);
  k_agg1<<<(N_NODES + 255) / 256, 256, 0, stream>>>(h1, rowptr, col, dinv, b_out, svec);

  // 6. fc1 (38332 -> 128) + relu
  k_zero_f32<<<1, 128, 0, stream>>>(hpre, 128);
  k_fc1<<<(N_NODES + 511) / 512, 128, 0, stream>>>(svec, fc1_W, hpre);
  k_fc1_act<<<1, 128, 0, stream>>>(hpre, fc1_b, hvec);

  // 7. fc2 (128 -> 38333) + relu
  k_fc2<<<(POI_LEN + 255) / 256, 256, 0, stream>>>(hvec, fc2_W, fc2_b, out);
}
